// ScannedRNN_4947802325200
// MI455X (gfx1250) — matrix operand order repaired and checked by reading, not compiled, not run on hardware
//
#include <hip/hip_runtime.h>
#include <hip/hip_bf16.h>

typedef __attribute__((ext_vector_type(16))) __bf16 v16bf;
typedef __attribute__((ext_vector_type(8)))  float  v8f;

#define T_STEPS 256
#define B_SZ    256
#define D_SZ    512
#define H_SZ    512
#define N3H     1536
#define KT_D    16   // D/32 k-chunks
#define KT_H    16   // H/32 k-chunks
#define LDS_STRIDE 520  // H + 8 pad (u16 units)

// ---------- helpers ----------
__device__ __forceinline__ unsigned short f2bf(float f) {
  union { float f; unsigned u; } v; v.f = f;
  unsigned r = v.u + 0x7FFFu + ((v.u >> 16) & 1u);   // round-to-nearest-even
  return (unsigned short)(r >> 16);
}
__device__ __forceinline__ float fsigmoid(float x) {
  return 1.0f / (1.0f + __expf(-x));
}
__device__ __forceinline__ float ftanh_fast(float x) {
  float e = __expf(2.0f * x);
  return 1.0f - 2.0f / (e + 1.0f);   // safe at +/-inf
}

// ---------- prep: stage W (f32, KxN row-major) into per-lane WMMA B layout ----------
// B tile 32x16 bf16: lane l holds N = l&15, K = (e<8 ? e : 16+(e-8)) + 8*(l>>4)
// dst element index: ((nt*KT + kt)*32 + lane)*16 + e
__global__ void stage_w_kernel(const float* __restrict__ W, unsigned short* __restrict__ dst,
                               int N, int K, int total) {
  int idx = blockIdx.x * blockDim.x + threadIdx.x;
  if (idx >= total) return;
  int e    = idx & 15;
  int lane = (idx >> 4) & 31;
  int rest = idx >> 9;
  int KT   = K >> 5;
  int kt   = rest % KT;
  // nt = rest / KT (implicit in idx layout)
  int kk = (kt << 5) + ((lane >> 4) << 3) + ((e < 8) ? e : (16 + (e - 8)));
  int nn = (((rest / KT) << 4)) + (lane & 15);
  dst[idx] = f2bf(W[(size_t)kk * N + nn]);
}

// ---------- prep: stage ins (f32, MxK row-major) into per-lane WMMA A layout ----------
// A tile 16x32 bf16 (ISA 7.12.2): lane l, M = l&15; VGPR j: K = (j<4 ? 2j : 16+2(j-4)) + 8*(l>>4)
__global__ void stage_a_kernel(const float* __restrict__ A, unsigned short* __restrict__ dst,
                               int K, int total) {
  int idx = blockIdx.x * blockDim.x + threadIdx.x;
  if (idx >= total) return;
  int e    = idx & 15;
  int lane = (idx >> 4) & 31;
  int rest = idx >> 9;
  int KT   = K >> 5;
  int kt   = rest % KT;
  int mt   = rest / KT;
  int j     = e >> 1;
  int khalf = (lane >> 4) << 3;
  int kk = (kt << 5) + ((j < 4) ? (2 * j + khalf) : (16 + 2 * (j - 4) + khalf)) + (e & 1);
  int mm = (mt << 4) + (lane & 15);
  dst[idx] = f2bf(A[(size_t)mm * K + kk]);
}

// ---------- phase 1: xi = ins @ Wi + bi  (65536 x 1536 x 512, bf16 WMMA) ----------
// WG = 256 threads (8 waves); WG tile 128x128; wave tile 32x64 (2x4 16x16 tiles)
__global__ __launch_bounds__(256) void xi_gemm_kernel(
    const unsigned short* __restrict__ Abits, const unsigned short* __restrict__ Bbits,
    const float* __restrict__ bi, float* __restrict__ xi) {
  const v16bf* Ap = (const v16bf*)Abits;
  const v16bf* Bp = (const v16bf*)Bbits;
  int lane = threadIdx.x & 31;
  int wave = threadIdx.x >> 5;
  int wr = wave >> 1;                         // 0..3
  int wc = wave & 1;                          // 0..1
  int mt0 = blockIdx.x * 8 + wr * 2;          // 16-row tile index
  int nt0 = blockIdx.y * 8 + wc * 4;          // 16-col tile index

  v8f acc[2][4] = {};

  #pragma unroll 4
  for (int kt = 0; kt < KT_D; ++kt) {
    v16bf a0 = Ap[(((size_t)mt0       * KT_D + kt) << 5) + lane];
    v16bf a1 = Ap[(((size_t)(mt0 + 1) * KT_D + kt) << 5) + lane];
    v16bf b[4];
    #pragma unroll
    for (int c = 0; c < 4; ++c)
      b[c] = Bp[(((size_t)(nt0 + c) * KT_D + kt) << 5) + lane];
    #pragma unroll
    for (int c = 0; c < 4; ++c) {
      acc[0][c] = __builtin_amdgcn_wmma_f32_16x16x32_bf16(false, a0, false, b[c],
                                                          (short)0, acc[0][c], false, false);
      acc[1][c] = __builtin_amdgcn_wmma_f32_16x16x32_bf16(false, a1, false, b[c],
                                                          (short)0, acc[1][c], false, false);
    }
  }

  int colo = lane & 15;
  int rowo = (lane >> 4) << 3;                // C layout: VGPR i -> row i + rowo
  #pragma unroll
  for (int r = 0; r < 2; ++r) {
    #pragma unroll
    for (int c = 0; c < 4; ++c) {
      int colg = ((nt0 + c) << 4) + colo;
      float bias = bi[colg];
      #pragma unroll
      for (int i = 0; i < 8; ++i) {
        int rowg = ((mt0 + r) << 4) + rowo + i;
        xi[(size_t)rowg * N3H + colg] = acc[r][c][i] + bias;
      }
    }
  }
}

// ---------- phase 2: persistent GRU scan ----------
// 16 WGs x 1024 thr. WG owns 16 batch rows for ALL timesteps (recurrence is
// independent per batch row -> zero cross-WG communication). Wave w owns
// hidden columns [16w,16w+16): computes hr/hz/hn tiles and the h_new slice.
// h kept in f32 registers; bf16 copy in LDS is the shared WMMA A-operand.
__global__ __launch_bounds__(1024, 1) void gru_scan_kernel(
    const float* __restrict__ xi, const unsigned char* __restrict__ resets,
    const float* __restrict__ h0, const unsigned short* __restrict__ WhBits,
    const float* __restrict__ bh_n, float* __restrict__ out) {
  __shared__ unsigned short hbf[16 * LDS_STRIDE];
  const v16bf* Bp = (const v16bf*)WhBits;
  float* hfin = out;
  float* ys   = out + (size_t)B_SZ * H_SZ;

  int lane = threadIdx.x & 31;
  int wave = threadIdx.x >> 5;                // 0..31
  int rowbase = blockIdx.x << 4;
  int colo = lane & 15;
  int rowo = (lane >> 4) << 3;
  int col  = (wave << 4) + colo;              // 0..511
  int khalf = (lane >> 4) << 3;

  float h[8];
  #pragma unroll
  for (int i = 0; i < 8; ++i)
    h[i] = h0[(size_t)(rowbase + rowo + i) * H_SZ + col];
  float bhn = bh_n[col];

  for (int t = 0; t < T_STEPS; ++t) {
    // episode reset masking (per batch row)
    const unsigned char* rs = resets + (size_t)t * B_SZ + rowbase + rowo;
    #pragma unroll
    for (int i = 0; i < 8; ++i)
      if (rs[i]) h[i] = 0.0f;

    __syncthreads();   // previous step's LDS A reads complete
    #pragma unroll
    for (int i = 0; i < 8; ++i)
      hbf[(rowo + i) * LDS_STRIDE + col] = f2bf(h[i]);
    __syncthreads();   // h tile visible to all waves

    // hh = h @ Wh : wave computes its r/z/n 16x16 tiles, K = 512
    v8f ar = {}, az = {}, an = {};
    const unsigned* hrow = (const unsigned*)(hbf + (lane & 15) * LDS_STRIDE);
    #pragma unroll
    for (int kt = 0; kt < KT_H; ++kt) {
      union { v16bf v; unsigned u[8]; } aun;
      int k0 = kt << 5;
      #pragma unroll
      for (int j = 0; j < 8; ++j) {
        int kk = (j < 4) ? (2 * j + khalf) : (16 + 2 * (j - 4) + khalf);
        aun.u[j] = hrow[(k0 + kk) >> 1];
      }
      v16bf br = Bp[(((size_t)(wave)      * KT_H + kt) << 5) + lane];
      v16bf bz = Bp[(((size_t)(wave + 32) * KT_H + kt) << 5) + lane];
      v16bf bn = Bp[(((size_t)(wave + 64) * KT_H + kt) << 5) + lane];
      ar = __builtin_amdgcn_wmma_f32_16x16x32_bf16(false, aun.v, false, br, (short)0, ar, false, false);
      az = __builtin_amdgcn_wmma_f32_16x16x32_bf16(false, aun.v, false, bz, (short)0, az, false, false);
      an = __builtin_amdgcn_wmma_f32_16x16x32_bf16(false, aun.v, false, bn, (short)0, an, false, false);
    }

    // gates + state update + ys store (f32)
    size_t xirow = ((size_t)t * B_SZ + rowbase + rowo) * N3H;
    size_t ysrow = ((size_t)t * B_SZ + rowbase + rowo) * H_SZ;
    #pragma unroll
    for (int i = 0; i < 8; ++i) {
      size_t xb = xirow + (size_t)i * N3H;
      float xr = xi[xb + col];
      float xz = xi[xb + 512 + col];
      float xn = xi[xb + 1024 + col];
      float r = fsigmoid(xr + ar[i]);
      float z = fsigmoid(xz + az[i]);
      float n = ftanh_fast(xn + r * (an[i] + bhn));
      h[i] = (1.0f - z) * n + z * h[i];
      ys[ysrow + (size_t)i * H_SZ + col] = h[i];
    }
  }

  #pragma unroll
  for (int i = 0; i < 8; ++i)
    hfin[(size_t)(rowbase + rowo + i) * H_SZ + col] = h[i];
}

// ---------- launch ----------
extern "C" void kernel_launch(void* const* d_in, const int* in_sizes, int n_in,
                              void* d_out, int out_size, void* d_ws, size_t ws_size,
                              hipStream_t stream) {
  const float*         ins    = (const float*)d_in[0];
  const unsigned char* resets = (const unsigned char*)d_in[1];  // jax bool_ = 1 byte
  const float*         h0     = (const float*)d_in[2];
  const float*         Wi     = (const float*)d_in[3];
  const float*         bi     = (const float*)d_in[4];
  const float*         Wh     = (const float*)d_in[5];
  const float*         bh_n   = (const float*)d_in[6];
  (void)in_sizes; (void)n_in; (void)out_size; (void)ws_size;

  // workspace layout
  char* ws = (char*)d_ws;
  const size_t SZ_A  = (size_t)T_STEPS * B_SZ * D_SZ * 2;  // 67,108,864 B (staged ins bf16)
  const size_t SZ_W  = (size_t)D_SZ * N3H * 2;             //  1,572,864 B (staged W bf16)
  unsigned short* stA  = (unsigned short*)(ws);
  unsigned short* stWi = (unsigned short*)(ws + SZ_A);
  unsigned short* stWh = (unsigned short*)(ws + SZ_A + SZ_W);
  float*          xi   = (float*)(ws + SZ_A + 2 * SZ_W);   // 402,653,184 B (f32 xi)

  // prep: bf16 conversion + WMMA operand staging
  int totalA = T_STEPS * B_SZ * D_SZ;       // 33,554,432
  stage_a_kernel<<<totalA / 256, 256, 0, stream>>>(ins, stA, D_SZ, totalA);
  int totalW = D_SZ * N3H;                  // 786,432
  stage_w_kernel<<<totalW / 256, 256, 0, stream>>>(Wi, stWi, N3H, D_SZ, totalW);
  stage_w_kernel<<<totalW / 256, 256, 0, stream>>>(Wh, stWh, N3H, H_SZ, totalW);

  // phase 1: xi = ins @ Wi + bi
  dim3 g1((T_STEPS * B_SZ) / 128, N3H / 128);   // 512 x 12
  xi_gemm_kernel<<<g1, 256, 0, stream>>>(stA, stWi, bi, xi);

  // phase 2: persistent GRU scan (one WG per 16 batch rows)
  gru_scan_kernel<<<B_SZ / 16, 1024, 0, stream>>>(xi, resets, h0, stWh, bh_n, (float*)d_out);
}
